// STULayer_71055938945776
// MI455X (gfx1250) — compile-verified, matrix operand order checked by reading, not run on hardware
//
#include <hip/hip_runtime.h>
#include <hip/hip_bf16.h>

// Problem constants (match reference)
#define BB 4
#define TT 4096
#define DD 256
#define KK 16
#define TC 32               // time-chunk rows = 2 WMMA M-tiles
#define NC (TT / TC)        // 128 chunks
#define KD (KK * DD)        // 4096 fused inner dim
#define SLICE 512           // A-tile slice of the inner dim (2 k's x 256 d)
#define NSLICE (KD / SLICE) // 8
#define APAD 8
#define ASTRIDE (SLICE + APAD)

typedef __bf16 bf16_t;
typedef bf16_t v16bf __attribute__((ext_vector_type(16)));
typedef bf16_t v8bf  __attribute__((ext_vector_type(8)));
typedef float  v8f   __attribute__((ext_vector_type(8)));

// ---------------------------------------------------------------------------
// Kernel 0: repack W (K,D,D) f32 -> Wt2 (e-major) bf16: Wt2[e][k*D+d] = W[k,e,d]
// so B-fragments are two contiguous 16B loads per lane.
// ---------------------------------------------------------------------------
__global__ void stu_wconv(const float* __restrict__ W, bf16_t* __restrict__ Wt2) {
    int idx = blockIdx.x * blockDim.x + threadIdx.x;   // over K*D*D
    int d = idx & (DD - 1);
    int e = (idx >> 8) & (DD - 1);
    int k = idx >> 16;
    Wt2[(size_t)e * KD + k * DD + d] = (bf16_t)W[idx];
}

// ---------------------------------------------------------------------------
// Kernel 1: per-chunk sums  S[b,c,k,d] = sum_{t in chunk c} phi[t,k]*x[b,t,d]
// ---------------------------------------------------------------------------
__global__ void stu_chunk_sums(const float* __restrict__ x,
                               const float* __restrict__ phi,
                               float* __restrict__ S) {
    __shared__ float xs[TC][DD];   // 32 KB
    __shared__ float ps[TC][KK];   //  2 KB
    int b = blockIdx.x / NC, c = blockIdx.x % NC;
    int t0 = c * TC;
    int tid = threadIdx.x; // 256
    for (int i = 0; i < (TC * DD) / 256; ++i) {
        int j = tid + i * 256;
        xs[j >> 8][j & 255] = x[((size_t)b * TT + t0 + (j >> 8)) * DD + (j & 255)];
    }
    for (int i = 0; i < (TC * KK) / 256; ++i) {
        int j = tid + i * 256;
        ps[j >> 4][j & 15] = phi[(size_t)(t0 + (j >> 4)) * KK + (j & 15)];
    }
    __syncthreads();
    int d = tid;
    for (int k = 0; k < KK; ++k) {
        float s = 0.f;
        #pragma unroll
        for (int t = 0; t < TC; ++t) s += ps[t][k] * xs[t][d];
        S[(((size_t)b * NC + c) * KK + k) * DD + d] = s;
    }
}

// ---------------------------------------------------------------------------
// Kernel 2: in-place exclusive scan over chunks -> S becomes carry[b,c,k,d]
// One thread per (b,k,d); 128 sequential chunk steps, fully coalesced.
// ---------------------------------------------------------------------------
__global__ void stu_chunk_scan(float* __restrict__ S) {
    int tid = blockIdx.x * blockDim.x + threadIdx.x;  // B*K*D = 16384
    int d = tid & (DD - 1);
    int k = (tid >> 8) & (KK - 1);
    int b = tid >> 12;
    float run = 0.f;
    for (int c = 0; c < NC; ++c) {
        size_t a = (((size_t)b * NC + c) * KK + k) * DD + d;
        float v = S[a];
        S[a] = run;
        run += v;
    }
}

// ---------------------------------------------------------------------------
// Main fused kernel: block = (b, 32-row chunk), 256 threads = 8 waves.
// Build A[t][k*D+d] = bf16(phi[t0+t,k]*(carry + local cumsum)) slice-by-slice
// in LDS; each wave computes a 32x32 output tile as 2x2 WMMA tiles:
//   A fragments (LDS) reused across N, B fragments (global Wt2) reused
//   across M, 4 independent accumulator chains for XDL pipelining.
// ---------------------------------------------------------------------------
__global__ __launch_bounds__(256) void stu_main(
    const float* __restrict__ x, const float* __restrict__ phi,
    const float* __restrict__ carry, const bf16_t* __restrict__ Wt2,
    float* __restrict__ out) {
    __shared__ bf16_t A[TC][ASTRIDE];   // 32 x 520 bf16 = 33.3 KB
    __shared__ float  ps[TC][KK];       //  2 KB

    int b = blockIdx.x / NC, c = blockIdx.x % NC;
    int t0 = c * TC;
    int tid  = threadIdx.x;
    int lane = tid & 31;
    int wave = tid >> 5;          // 0..7
    int e0   = wave * 32;         // 32 output columns per wave (2 N-tiles)

    for (int i = 0; i < (TC * KK) / 256; ++i) {
        int j = tid + i * 256;
        ps[j >> 4][j & 15] = phi[(size_t)(t0 + (j >> 4)) * KK + (j & 15)];
    }

    v8f acc00 = {}, acc01 = {}, acc10 = {}, acc11 = {};
    const int m     = lane & 15;          // row/col within a 16x16 tile
    const int khalf = (lane >> 4) * 8;    // ISA 16-bit A/B lane-half K offset
    const bf16_t* Brow0 = Wt2 + (size_t)(e0 + m)      * KD + khalf;
    const bf16_t* Brow1 = Wt2 + (size_t)(e0 + 16 + m) * KD + khalf;

    for (int s = 0; s < NSLICE; ++s) {
        __syncthreads();  // also covers initial ps fill
        // ---- build A slice: 512 (k,d) pairs, 2 per thread, 32-row cumsum ----
        #pragma unroll
        for (int i = 0; i < 2; ++i) {
            int kdl = tid + i * 256;              // 0..511 within slice
            int kd  = s * SLICE + kdl;
            int k   = kd >> 8;
            int d   = kd & 255;
            float run = carry[(((size_t)b * NC + c) * KK + k) * DD + d];
            const float* xp = x + ((size_t)b * TT + t0) * DD + d;
            #pragma unroll
            for (int t = 0; t < TC; ++t) {
                float p = ps[t][k];
                run += p * xp[t * DD];            // coalesced (lane <-> d)
                A[t][kdl] = (bf16_t)(p * run);
            }
        }
        __syncthreads();
        // ---- consume: 16 WMMA K-steps of 32; 2x2 register tile ----
        const bf16_t* B0 = Brow0 + s * SLICE;
        const bf16_t* B1 = Brow1 + s * SLICE;
        #pragma unroll 2
        for (int kk = 0; kk < SLICE / 32; ++kk) {
            __builtin_prefetch(B0 + (kk + 8) * 32, 0, 0);  // global_prefetch_b8
            __builtin_prefetch(B1 + (kk + 8) * 32, 0, 0);
            v16bf a0, a1, b0, b1;
            // A: M-tile 0 (rows 0-15) and M-tile 1 (rows 16-31)
            *(v8bf*)&a0       = *(const v8bf*)&A[m][kk * 32 + khalf];
            *((v8bf*)&a0 + 1) = *(const v8bf*)&A[m][kk * 32 + khalf + 16];
            *(v8bf*)&a1       = *(const v8bf*)&A[m + 16][kk * 32 + khalf];
            *((v8bf*)&a1 + 1) = *(const v8bf*)&A[m + 16][kk * 32 + khalf + 16];
            // B: N-tile 0 and N-tile 1 (streamed from L2-resident Wt2)
            *(v8bf*)&b0       = *(const v8bf*)(B0 + kk * 32);
            *((v8bf*)&b0 + 1) = *(const v8bf*)(B0 + kk * 32 + 16);
            *(v8bf*)&b1       = *(const v8bf*)(B1 + kk * 32);
            *((v8bf*)&b1 + 1) = *(const v8bf*)(B1 + kk * 32 + 16);
            // 4 independent accumulator chains
            acc00 = __builtin_amdgcn_wmma_f32_16x16x32_bf16(false, a0, false, b0,
                                                            (short)0, acc00, false, false);
            acc01 = __builtin_amdgcn_wmma_f32_16x16x32_bf16(false, a0, false, b1,
                                                            (short)0, acc01, false, false);
            acc10 = __builtin_amdgcn_wmma_f32_16x16x32_bf16(false, a1, false, b0,
                                                            (short)0, acc10, false, false);
            acc11 = __builtin_amdgcn_wmma_f32_16x16x32_bf16(false, a1, false, b1,
                                                            (short)0, acc11, false, false);
        }
    }

    // D layout: VGPR i, lanes 0-15 -> M=i, lanes 16-31 -> M=i+8; N = lane&15
    const int mhi = (lane >> 4) * 8;
    float* outb = out + ((size_t)b * TT + t0) * DD;
    #pragma unroll
    for (int i = 0; i < 8; ++i) {
        int mm = i + mhi;
        outb[(size_t)(mm)      * DD + e0 + m]      = acc00[i];
        outb[(size_t)(mm)      * DD + e0 + 16 + m] = acc01[i];
        outb[(size_t)(mm + 16) * DD + e0 + m]      = acc10[i];
        outb[(size_t)(mm + 16) * DD + e0 + 16 + m] = acc11[i];
    }
}

// ---------------------------------------------------------------------------
extern "C" void kernel_launch(void* const* d_in, const int* in_sizes, int n_in,
                              void* d_out, int out_size, void* d_ws, size_t ws_size,
                              hipStream_t stream) {
    (void)in_sizes; (void)n_in; (void)out_size; (void)ws_size;
    const float* x   = (const float*)d_in[0];  // (B,T,D)
    const float* W   = (const float*)d_in[1];  // (K,D,D)
    const float* phi = (const float*)d_in[2];  // (T,K)
    float* out = (float*)d_out;                // (B,T,D)

    char* ws = (char*)d_ws;
    bf16_t* Wt2 = (bf16_t*)ws;                                     // 2 MB
    float*  S   = (float*)(ws + (size_t)KD * DD * sizeof(bf16_t)); // 8.4 MB (becomes carry)

    stu_wconv     <<<(KK * DD * DD) / 256,  256, 0, stream>>>(W, Wt2);
    stu_chunk_sums<<<BB * NC,               256, 0, stream>>>(x, phi, S);
    stu_chunk_scan<<<(BB * KK * DD) / 256,  256, 0, stream>>>(S);
    stu_main      <<<BB * NC,               256, 0, stream>>>(x, phi, S, Wt2, out);
}
